// TransformerVAE_76072460747455
// MI455X (gfx1250) — matrix operand order repaired and checked by reading, not compiled, not run on hardware
//
#include <hip/hip_runtime.h>
#include <math.h>

// ---------------------------------------------------------------------------
// TransformerVAE forward for MI455X (gfx1250, wave32, WMMA).
//
// Input leaf order (jax pytree: dict keys sorted alphabetically per level):
//  0: codebook (1024,1024)
//  1: dec deconv_b(3)  2: dec deconv_w(3,64)
//  3..14 : dec g0  (2 blocks, d=64)   keys: b_fc,b_o,b_pr,b_qkv,ln1_b,ln1_g,
//  15..26: dec g1  (4 blocks, d=256)         ln2_b,ln2_g,w_fc,w_o,w_pr,w_qkv
//  27..38: dec main(5 blocks, d=1024)
//  39: dec pos0(1,64,1024) 40: dec pos1(1,256,256) 41: dec pos2(1,1024,64)
//  42: enc conv_b(64) 43: enc conv_w(64,3)
//  44..55: enc g0 (2, d=64)  56..67: enc g1 (4, d=256)  68..79: enc main (5, d=1024)
//  80: enc pos0(1,1024,64) 81: enc pos1(1,256,256) 82: enc pos2(1,64,1024)
//  83: x (16,3,32,32)
//
// d_out: [0..1023] encoding_ids (as float), [1024] loss, [1025..50176] dec_out.
// Workspace: 12M floats (48 MB).
// ---------------------------------------------------------------------------

typedef __bf16 bf16;
typedef __attribute__((ext_vector_type(16))) __bf16 v16bf;
typedef __attribute__((ext_vector_type(8)))  float  v8f;

// ------------------------------ WMMA GEMM ----------------------------------
// C[M,N] = act( A[M,K] @ B^T + bias ) + res     (B stored [N,K] when !B_KMAJOR,
//                                                [K,N] when B_KMAJOR)
// Block = 128 threads (4 waves) computing a 128x64 C tile. Wave w owns rows
// [w*32, w*32+32): 2 A-fragments x 4 B-fragments = 8 WMMAs per 32-wide K-step.
// Double-buffered LDS + register prefetch: the next tile's global loads are in
// flight while the current tile's WMMAs execute; one barrier per K-step.
template <bool B_KMAJOR, bool HAS_BIAS, bool GELU, bool HAS_RES>
__global__ __launch_bounds__(128) void gemm_wmma_kernel(
    const float* __restrict__ A, const float* __restrict__ Bm,
    const float* __restrict__ bias, const float* __restrict__ res,
    float* __restrict__ C, int M, int N, int K) {
  __shared__ bf16 As[2][128][40];  // padded rows, 16B-aligned chunks
  __shared__ bf16 Bs[2][64][40];   // stored [n][k] for both B layouts

  const int m0   = blockIdx.y * 128;
  const int n0   = blockIdx.x * 64;
  const int tid  = threadIdx.x;
  const int wave = tid >> 5;
  const int lane = tid & 31;
  const int half = lane >> 4;  // 0: lanes 0-15, 1: lanes 16-31
  const int l16  = lane & 15;

  v8f   acc[2][4] = {};
  float ra[32];  // A prefetch: row tid, 32 consecutive K
  float rb[16];  // B prefetch

  auto load_regs = [&](int k0) {
    const float* srcA = A + (size_t)(m0 + tid) * K + k0;
#pragma unroll
    for (int j = 0; j < 32; ++j) ra[j] = srcA[j];
    if (B_KMAJOR) {  // Bm[k][n]
      int k = tid >> 2, q = tid & 3;
      const float* srcB = Bm + (size_t)(k0 + k) * N + n0 + q * 16;
#pragma unroll
      for (int j = 0; j < 16; ++j) rb[j] = srcB[j];
    } else {  // Bm[n][k]
      int rr = tid >> 1, hh = tid & 1;
      const float* srcB = Bm + (size_t)(n0 + rr) * K + k0 + hh * 16;
#pragma unroll
      for (int j = 0; j < 16; ++j) rb[j] = srcB[j];
    }
  };
  auto store_lds = [&](int buf) {
    bf16* dA = &As[buf][tid][0];
#pragma unroll
    for (int j = 0; j < 32; ++j) dA[j] = (bf16)ra[j];
    if (B_KMAJOR) {  // transpose into Bs[n][k]
      int k = tid >> 2, q = tid & 3;
#pragma unroll
      for (int j = 0; j < 16; ++j) Bs[buf][q * 16 + j][k] = (bf16)rb[j];
    } else {
      int rr = tid >> 1, hh = tid & 1;
      bf16* dB = &Bs[buf][rr][hh * 16];
#pragma unroll
      for (int j = 0; j < 16; ++j) dB[j] = (bf16)rb[j];
    }
  };

  load_regs(0);
  store_lds(0);
  __syncthreads();

  const int nsteps = K >> 5;
  for (int s = 0; s < nsteps; ++s) {
    const int  p    = s & 1;
    const bool more = (s + 1 < nsteps);
    if (more) load_regs((s + 1) << 5);  // global loads overlap the WMMAs below

    // A fragments (16x32, ISA 7.12.2): lane row = l16, elems 0..7 ->
    // K=half*8+e, elems 8..15 -> K=16+half*8+e.
    v16bf afrag[2];
#pragma unroll
    for (int mt = 0; mt < 2; ++mt) {
      const bf16* rowp = &As[p][wave * 32 + mt * 16 + l16][0];
#pragma unroll
      for (int e = 0; e < 8; ++e) {
        afrag[mt][e]     = rowp[half * 8 + e];
        afrag[mt][e + 8] = rowp[16 + half * 8 + e];
      }
    }
#pragma unroll
    for (int t = 0; t < 4; ++t) {
      // B fragment (32x16): lane col = l16, elems 0..7 -> K=half*8+e, 8..15 -> K=16+half*8+e.
      const bf16* rowp = &Bs[p][t * 16 + l16][0];
      v16bf b;
#pragma unroll
      for (int e = 0; e < 16; ++e) b[e] = rowp[half * 8 + e + (e >= 8 ? 8 : 0)];
      acc[0][t] = __builtin_amdgcn_wmma_f32_16x16x32_bf16(
          false, afrag[0], false, b, (short)0, acc[0][t], false, false);
      acc[1][t] = __builtin_amdgcn_wmma_f32_16x16x32_bf16(
          false, afrag[1], false, b, (short)0, acc[1][t], false, false);
    }
    if (more) store_lds(p ^ 1);
    __syncthreads();  // separates this step's writes from next step's reads,
                      // and previous reads from next writes (double buffer)
  }

  // Epilogue. C layout: VGPR r, lanes 0-15 -> M=r, lanes 16-31 -> M=8+r, N=l16.
#pragma unroll
  for (int t = 0; t < 4; ++t) {
    const int   n  = n0 + t * 16 + l16;
    const float bv = HAS_BIAS ? bias[n] : 0.f;
#pragma unroll
    for (int mt = 0; mt < 2; ++mt) {
#pragma unroll
      for (int r = 0; r < 8; ++r) {
        const int m = m0 + wave * 32 + mt * 16 + half * 8 + r;
        float v = acc[mt][t][r] + bv;
        if (GELU) v = v * (1.0f / (1.0f + __expf(-1.702f * v)));  // quickGELU
        if (HAS_RES) v += res[(size_t)m * N + n];
        C[(size_t)m * N + n] = v;
      }
    }
  }
}

// ------------------------------ LayerNorm ----------------------------------
__global__ __launch_bounds__(256) void ln_kernel(
    const float* __restrict__ x, const float* __restrict__ g,
    const float* __restrict__ bta, float* __restrict__ y, int M, int d) {
  const int row  = blockIdx.x * 8 + (threadIdx.x >> 5);
  const int lane = threadIdx.x & 31;
  if (row >= M) return;
  const float* xr = x + (size_t)row * d;
  float s = 0.f, s2 = 0.f;
  for (int e = lane; e < d; e += 32) {
    float v = xr[e];
    s += v;
    s2 += v * v;
  }
#pragma unroll
  for (int o = 16; o > 0; o >>= 1) {
    s += __shfl_xor(s, o, 32);
    s2 += __shfl_xor(s2, o, 32);
  }
  const float m   = s / d;
  const float var = s2 / d - m * m;
  const float rs  = rsqrtf(var + 1e-5f);
  float* yr = y + (size_t)row * d;
  for (int e = lane; e < d; e += 32) yr[e] = (xr[e] - m) * rs * g[e] + bta[e];
}

// ------------------------------ Attention ----------------------------------
// qkv: (B, Ntok, 3d) with q|k|v concatenated. One block per (b, head),
// 8 waves; each wave owns queries q = wave, wave+8, ...  hd = d/8 (8..128).
__global__ __launch_bounds__(256) void attn_kernel(
    const float* __restrict__ qkv, float* __restrict__ out, int Ntok, int d,
    int nh) {
  extern __shared__ float sc[];  // 8 * Ntok score rows
  const int hd   = d / nh;
  const int b    = blockIdx.x / nh;
  const int h    = blockIdx.x % nh;
  const int wave = threadIdx.x >> 5;
  const int lane = threadIdx.x & 31;
  float* mysc = sc + wave * Ntok;
  const float scale   = rsqrtf((float)hd);
  const int   nchunks = (hd + 31) / 32;  // <= 4
  const float* base   = qkv + (size_t)b * Ntok * 3 * d;

  for (int q = wave; q < Ntok; q += 8) {
    const float* qrow = base + (size_t)q * 3 * d + h * hd;
    float qreg[4];
#pragma unroll
    for (int c = 0; c < 4; ++c) {
      int e = c * 32 + lane;
      qreg[c] = (c < nchunks && e < hd) ? qrow[e] : 0.f;
    }
    float mx = -3.4e38f;
    for (int t = 0; t < Ntok; ++t) {
      const float* krow = base + (size_t)t * 3 * d + d + h * hd;
      float dot = 0.f;
#pragma unroll
      for (int c = 0; c < 4; ++c) {
        int e = c * 32 + lane;
        if (c < nchunks && e < hd) dot += qreg[c] * krow[e];
      }
#pragma unroll
      for (int o = 16; o > 0; o >>= 1) dot += __shfl_xor(dot, o, 32);
      dot *= scale;
      if (lane == 0) mysc[t] = dot;
      mx = fmaxf(mx, dot);
    }
    float ps = 0.f;
    for (int t = lane; t < Ntok; t += 32) {
      float e = __expf(mysc[t] - mx);
      mysc[t] = e;
      ps += e;
    }
#pragma unroll
    for (int o = 16; o > 0; o >>= 1) ps += __shfl_xor(ps, o, 32);
    const float inv = 1.f / ps;
    for (int t = lane; t < Ntok; t += 32) mysc[t] *= inv;

    float accv[4] = {0.f, 0.f, 0.f, 0.f};
    for (int t = 0; t < Ntok; ++t) {
      const float s = mysc[t];
      const float* vrow = base + (size_t)t * 3 * d + 2 * d + h * hd;
#pragma unroll
      for (int c = 0; c < 4; ++c) {
        int e = c * 32 + lane;
        if (c < nchunks && e < hd) accv[c] += s * vrow[e];
      }
    }
    float* orow = out + ((size_t)b * Ntok + q) * d + h * hd;
#pragma unroll
    for (int c = 0; c < 4; ++c) {
      int e = c * 32 + lane;
      if (c < nchunks && e < hd) orow[e] = accv[c];
    }
  }
}

// ------------------------------ Quantize -----------------------------------
// softmax over d=1024, argmax (first max), scatter 1.0 at argmax, emit id.
__global__ void quantize_kernel(const float* __restrict__ x,
                                float* __restrict__ hard,
                                float* __restrict__ ids, int d) {
  const int row  = blockIdx.x;
  const int lane = threadIdx.x;
  const float* xr = x + (size_t)row * d;
  float mx = -3.4e38f;
  int   mi = 0x7fffffff;
  for (int e = lane; e < d; e += 32) {
    float v = xr[e];
    if (v > mx) { mx = v; mi = e; }
  }
#pragma unroll
  for (int o = 16; o > 0; o >>= 1) {
    float ov = __shfl_xor(mx, o, 32);
    int   oi = __shfl_xor(mi, o, 32);
    if (ov > mx || (ov == mx && oi < mi)) { mx = ov; mi = oi; }
  }
  float ps = 0.f;
  for (int e = lane; e < d; e += 32) ps += __expf(xr[e] - mx);
#pragma unroll
  for (int o = 16; o > 0; o >>= 1) ps += __shfl_xor(ps, o, 32);
  const float inv = 1.f / ps;
  float* hr = hard + (size_t)row * d;
  for (int e = lane; e < d; e += 32)
    hr[e] = (e == mi) ? 1.0f : __expf(xr[e] - mx) * inv;
  if (lane == 0) ids[row] = (float)mi;
}

// --------------------------- Elementwise kernels ---------------------------
__global__ void conv_in_kernel(const float* __restrict__ x,
                               const float* __restrict__ cw,
                               const float* __restrict__ cb,
                               const float* __restrict__ pos,
                               float* __restrict__ out) {
  int idx = blockIdx.x * 256 + threadIdx.x;  // (b, n, o): 16*1024*64
  if (idx >= 16 * 1024 * 64) return;
  int o = idx & 63;
  int n = (idx >> 6) & 1023;
  int b = idx >> 16;
  float v = cb[o];
#pragma unroll
  for (int c = 0; c < 3; ++c)
    v += x[(size_t)b * 3072 + c * 1024 + n] * cw[o * 3 + c];
  out[idx] = v + pos[n * 64 + o];
}

// raw-reshape "pixel unshuffle": out[b][n'][c'] with i = c'*N2+n' flattened.
__global__ void pixel_down_kernel(const float* __restrict__ X,
                                  const float* __restrict__ pos,
                                  float* __restrict__ out, int H, int W, int C) {
  const int HW = H * W, N2 = HW / 4, C2 = 4 * C, per_b = N2 * C2;
  int idx = blockIdx.x * 256 + threadIdx.x;
  if (idx >= 16 * per_b) return;
  int b = idx / per_b, j = idx % per_b;
  int n2 = j / C2, c2 = j % C2;
  int i = c2 * N2 + n2;
  int c = i / HW, rem = i % HW;
  out[idx] = X[((size_t)b * HW + rem) * C + c] + pos[j];
}

// true PixelShuffle(r=2)
__global__ void pixel_up_kernel(const float* __restrict__ X,
                                const float* __restrict__ pos,
                                float* __restrict__ out, int H, int W, int C) {
  const int HW = H * W, C2 = C / 4, W2 = 2 * W, N2 = 4 * HW, per_b = N2 * C2;
  int idx = blockIdx.x * 256 + threadIdx.x;
  if (idx >= 16 * per_b) return;
  int b = idx / per_b, j = idx % per_b;
  int n2 = j / C2, c2 = j % C2;
  int h2 = n2 / W2, w2 = n2 % W2;
  int h = h2 >> 1, i = h2 & 1, w = w2 >> 1, jj = w2 & 1;
  int c = c2 * 4 + i * 2 + jj;
  out[idx] = X[((size_t)b * HW + h * W + w) * C + c] + pos[j];
}

__global__ void add_pos_kernel(float* __restrict__ x,
                               const float* __restrict__ pos, int M, int Ntok,
                               int d) {
  int idx = blockIdx.x * 256 + threadIdx.x;
  if (idx >= M * d) return;
  int row = idx / d, c = idx % d;
  x[idx] += pos[(row % Ntok) * d + c];
}

__global__ void zero1_kernel(float* p) { *p = 0.f; }

__global__ __launch_bounds__(256) void deconv_loss_kernel(
    const float* __restrict__ act, const float* __restrict__ dw,
    const float* __restrict__ db, const float* __restrict__ xin,
    float* __restrict__ dec_out, float* __restrict__ loss, int total) {
  __shared__ float red[256];
  int idx = blockIdx.x * 256 + threadIdx.x;  // (b, o, n) flat
  float contrib = 0.f;
  if (idx < total) {
    int n = idx & 1023;
    int o = (idx >> 10) % 3;
    int b = idx / 3072;
    const float* tr = act + ((size_t)b * 1024 + n) * 64;
    float v = db[o];
#pragma unroll 16
    for (int c = 0; c < 64; ++c) v += tr[c] * dw[o * 64 + c];
    dec_out[idx] = v;
    float df = v - xin[idx];
    contrib = df * df / (float)total;
  }
  red[threadIdx.x] = contrib;
  __syncthreads();
  for (int s = 128; s > 0; s >>= 1) {
    if ((int)threadIdx.x < s) red[threadIdx.x] += red[threadIdx.x + s];
    __syncthreads();
  }
  if (threadIdx.x == 0) atomicAdd(loss, red[0]);
}

// ------------------------------ Host driver --------------------------------
extern "C" void kernel_launch(void* const* d_in, const int* in_sizes, int n_in,
                              void* d_out, int out_size, void* d_ws,
                              size_t ws_size, hipStream_t stream) {
  (void)in_sizes; (void)n_in; (void)out_size; (void)ws_size;
  const float* const* in = (const float* const*)d_in;
  float* ws  = (float*)d_ws;
  float* out = (float*)d_out;

  const size_t MB = 1u << 20;  // 1M floats
  float* ACT0 = ws;
  float* ACT1 = ws + 1 * MB;
  float* Hb   = ws + 2 * MB;
  float* QKV  = ws + 3 * MB;   // 3M
  float* ATT  = ws + 6 * MB;   // 1M
  float* MLP  = ws + 7 * MB;   // 4M
  float* HARD = ws + 11 * MB;  // 1M

  // Dispatch: bias always present for block GEMMs; res => residual-add,
  // act => quickGELU. 128x64 tile: grid = (N/64, M/128).
  auto gemm = [&](const float* A, const float* Wt, const float* bias,
                  const float* res, float* C, int M, int N, int K, int act) {
    dim3 g((unsigned)(N / 64), (unsigned)(M / 128));
    if (res)
      gemm_wmma_kernel<false, true, false, true>
          <<<g, 128, 0, stream>>>(A, Wt, bias, res, C, M, N, K);
    else if (act)
      gemm_wmma_kernel<false, true, true, false>
          <<<g, 128, 0, stream>>>(A, Wt, bias, nullptr, C, M, N, K);
    else
      gemm_wmma_kernel<false, true, false, false>
          <<<g, 128, 0, stream>>>(A, Wt, bias, nullptr, C, M, N, K);
  };

  auto run_block = [&](float* x, int base, int blk, int d, int Ntok) {
    const int M = 16 * Ntok;
    const float* bfc  = in[base + 0] + (size_t)blk * 4 * d;
    const float* bo   = in[base + 1] + (size_t)blk * d;
    const float* bpr  = in[base + 2] + (size_t)blk * d;
    const float* bqkv = in[base + 3] + (size_t)blk * 3 * d;
    const float* ln1b = in[base + 4] + (size_t)blk * d;
    const float* ln1g = in[base + 5] + (size_t)blk * d;
    const float* ln2b = in[base + 6] + (size_t)blk * d;
    const float* ln2g = in[base + 7] + (size_t)blk * d;
    const float* wfc  = in[base + 8] + (size_t)blk * 4 * d * d;
    const float* wo   = in[base + 9] + (size_t)blk * d * d;
    const float* wpr  = in[base + 10] + (size_t)blk * 4 * d * d;
    const float* wqkv = in[base + 11] + (size_t)blk * 3 * d * d;

    ln_kernel<<<M / 8, 256, 0, stream>>>(x, ln1g, ln1b, Hb, M, d);
    gemm(Hb, wqkv, bqkv, nullptr, QKV, M, 3 * d, d, 0);
    attn_kernel<<<16 * 8, 256, 8 * Ntok * sizeof(float), stream>>>(QKV, ATT,
                                                                   Ntok, d, 8);
    gemm(ATT, wo, bo, x, x, M, d, d, 0);
    ln_kernel<<<M / 8, 256, 0, stream>>>(x, ln2g, ln2b, Hb, M, d);
    gemm(Hb, wfc, bfc, nullptr, MLP, M, 4 * d, d, 1);
    gemm(MLP, wpr, bpr, x, x, M, d, 4 * d, 0);
  };

  // ----------------- Encoder -----------------
  conv_in_kernel<<<(16 * 1024 * 64) / 256, 256, 0, stream>>>(
      in[83], in[43], in[42], in[80], ACT0);
  for (int i = 0; i < 2; ++i) run_block(ACT0, 44, i, 64, 1024);
  pixel_down_kernel<<<(16 * 256 * 256) / 256, 256, 0, stream>>>(ACT0, in[81],
                                                                ACT1, 32, 32, 64);
  for (int i = 0; i < 4; ++i) run_block(ACT1, 56, i, 256, 256);
  pixel_down_kernel<<<(16 * 64 * 1024) / 256, 256, 0, stream>>>(
      ACT1, in[82], ACT0, 16, 16, 256);
  for (int i = 0; i < 5; ++i) run_block(ACT0, 68, i, 1024, 64);

  // ----------------- Quantize -----------------
  quantize_kernel<<<1024, 32, 0, stream>>>(ACT0, HARD, out, 1024);
  {
    dim3 g(16, 8);  // hard(1024x1024) @ codebook(1024x1024), B is K-major
    gemm_wmma_kernel<true, false, false, false><<<g, 128, 0, stream>>>(
        HARD, in[0], nullptr, nullptr, ACT0, 1024, 1024, 1024);
  }
  add_pos_kernel<<<(1024 * 1024) / 256, 256, 0, stream>>>(ACT0, in[39], 1024,
                                                          64, 1024);

  // ----------------- Decoder -----------------
  for (int i = 0; i < 5; ++i) run_block(ACT0, 27, i, 1024, 64);
  pixel_up_kernel<<<(16 * 256 * 256) / 256, 256, 0, stream>>>(ACT0, in[40],
                                                              ACT1, 8, 8, 1024);
  for (int i = 0; i < 4; ++i) run_block(ACT1, 15, i, 256, 256);
  pixel_up_kernel<<<(16 * 1024 * 64) / 256, 256, 0, stream>>>(ACT1, in[41],
                                                              ACT0, 16, 16, 256);
  for (int i = 0; i < 2; ++i) run_block(ACT0, 3, i, 64, 1024);

  // ----------------- Deconv + loss -----------------
  zero1_kernel<<<1, 1, 0, stream>>>(out + 1024);
  deconv_loss_kernel<<<(49152 + 255) / 256, 256, 0, stream>>>(
      ACT0, in[2], in[1], in[83], out + 1025, out + 1024, 49152);
}
